// ConvGRU_21002390078206
// MI455X (gfx1250) — compile-verified, matrix operand order checked by reading, not run on hardware
//
#include <hip/hip_runtime.h>
#include <hip/hip_bf16.h>

typedef _Float16 f16;
typedef __attribute__((ext_vector_type(16))) _Float16 v16h;
typedef __attribute__((ext_vector_type(8)))  float    v8f;

#define DIM      64
#define DEPTHS   3
#define N_NODES  8192
#define N_EDGES  16384
#define N_GRAPHS 64

// packed A fragments per depth: 256 tiles * 4 kchunks * 32 lanes * 16 halves
#define PACKA_HALVES_PER_DEPTH (256 * 4 * 32 * 16)   // 524288

// ---------------------------------------------------------------------------
// Pack mlp2_w (f32 [3][4096][128]) into f16 WMMA A-fragment layout.
// A 16x32 f16 fragment (16x16x32 WMMA): lane<16 holds row M=lane, K slots
// {0..7,16..23}; lane>=16 holds row M=lane-16, K slots {8..15,24..31}.
// ---------------------------------------------------------------------------
__global__ void k_pack_mlp2(const float* __restrict__ mlp2_w,
                            f16* __restrict__ packA) {
    int idx = blockIdx.x * 256 + threadIdx.x;
    if (idx >= DEPTHS * PACKA_HALVES_PER_DEPTH) return;
    int d      = idx / PACKA_HALVES_PER_DEPTH;
    int rem    = idx - d * PACKA_HALVES_PER_DEPTH;
    int frag   = rem >> 9;          // fragment = 512 halves
    int within = rem & 511;
    int lane   = within >> 4;
    int s      = within & 15;
    int t      = frag >> 2;         // M tile (16 rows)
    int c      = frag & 3;          // K chunk (32)
    int row    = t * 16 + (lane & 15);
    int klocal = (lane < 16) ? ((s < 8) ? s : s + 8)
                             : ((s < 8) ? s + 8 : s + 16);
    int k = c * 32 + klocal;
    packA[idx] = (f16)mlp2_w[(size_t)d * 4096 * 128 + (size_t)row * 128 + k];
}

// ---------------------------------------------------------------------------
// Node encoder: x0 = relu(fc0(x)); h = GRU(x0, h0=0)  (gh == bhh since h0=0)
// ---------------------------------------------------------------------------
__global__ __launch_bounds__(64) void k_node0(
    const float* __restrict__ x, const float* __restrict__ fc0_w,
    const float* __restrict__ fc0_b, const float* __restrict__ wih,
    const float* __restrict__ bih, const float* __restrict__ bhh,
    float* __restrict__ h_out) {
    __shared__ float xl[40];
    __shared__ float x0[32];
    const int n = blockIdx.x, t = threadIdx.x;
    if (t < 40) xl[t] = x[n * 40 + t];
    __syncthreads();
    if (t < 32) {
        float a = fc0_b[t];
        #pragma unroll
        for (int k = 0; k < 40; ++k) a += xl[k] * fc0_w[t * 40 + k];
        x0[t] = a > 0.f ? a : 0.f;
    }
    __syncthreads();
    float gr = bih[t], gz = bih[64 + t], gn = bih[128 + t];
    #pragma unroll 4
    for (int k = 0; k < 32; ++k) {
        float xk = x0[k];
        gr += xk * wih[t * 32 + k];
        gz += xk * wih[(64 + t) * 32 + k];
        gn += xk * wih[(128 + t) * 32 + k];
    }
    float r  = 1.f / (1.f + __expf(-(gr + bhh[t])));
    float z  = 1.f / (1.f + __expf(-(gz + bhh[64 + t])));
    float nn = tanhf(gn + r * bhh[128 + t]);
    h_out[n * DIM + t] = (1.f - z) * nn;
}

// ---------------------------------------------------------------------------
// Fused edge kernel: edge MLP1 -> WMMA-generated We tiles -> contraction with
// h[src] + bias -> atomic scatter-add of messages into agg.
// Block = 128 threads (4 wave32), 64 edges per block. Wave w owns o-range
// [16w,16w+16) so message accumulation is race-free in registers.
// ---------------------------------------------------------------------------
__global__ __launch_bounds__(128) void k_edge(
    const f16*   __restrict__ packA,     // this depth, 524288 halves
    const float* __restrict__ mlp1_w,    // [128,10]
    const float* __restrict__ mlp1_b,    // [128]
    const float* __restrict__ mlp2_b,    // [4096]
    const float* __restrict__ edge_attr, // [E,10]
    const int*   __restrict__ ei,        // [2,E]
    const float* __restrict__ h_in,      // [N,64]
    float*       __restrict__ agg) {     // [N,64]
    __shared__ float attrs[64][10];
    __shared__ float hid[64][128];
    __shared__ float hsrc[64][64];

    const int tid  = threadIdx.x;
    const int lane = tid & 31;
    const int w    = tid >> 5;
    const int e0   = blockIdx.x * 64;

    for (int i = tid; i < 64 * 10; i += 128)
        attrs[i / 10][i % 10] = edge_attr[e0 * 10 + i];
    for (int i = tid; i < 64 * 64; i += 128) {
        int e = i >> 6, o = i & 63;
        int s = ei[e0 + e];
        hsrc[e][o] = h_in[s * DIM + o];
    }
    __syncthreads();
    for (int i = tid; i < 64 * 128; i += 128) {
        int e = i >> 7, c = i & 127;
        float a = mlp1_b[c];
        #pragma unroll
        for (int k = 0; k < 10; ++k) a += attrs[e][k] * mlp1_w[c * 10 + k];
        hid[e][c] = a > 0.f ? a : 0.f;
    }
    __syncthreads();

    // Pack B fragments (hid^T): lane gives N (edge), same K split as A.
    const int eL = lane & 15;
    v16h Bf[16];
    #pragma unroll
    for (int nt = 0; nt < 4; ++nt) {
        #pragma unroll
        for (int c = 0; c < 4; ++c) {
            v16h b;
            #pragma unroll
            for (int s = 0; s < 16; ++s) {
                int klocal = (lane < 16) ? ((s < 8) ? s : s + 8)
                                         : ((s < 8) ? s + 8 : s + 16);
                b[s] = (f16)hid[nt * 16 + eL][c * 32 + klocal];
            }
            Bf[nt * 4 + c] = b;
        }
    }

    float macc[4][8];
    #pragma unroll
    for (int nt = 0; nt < 4; ++nt)
        #pragma unroll
        for (int v = 0; v < 8; ++v) macc[nt][v] = 0.f;

    const v16h* Ap  = (const v16h*)packA;
    const int   mhi = (lane >= 16) ? 8 : 0;

    for (int j = 0; j < 64; ++j) {      // j == input channel i of We[i][o]
        const int t = j * 4 + w;        // M tile; o_base = 16*w, i = t>>2 = j
        v16h Af[4];
        #pragma unroll
        for (int c = 0; c < 4; ++c) Af[c] = Ap[(t * 4 + c) * 32 + lane];
        float bias[8];
        #pragma unroll
        for (int v = 0; v < 8; ++v) bias[v] = mlp2_b[t * 16 + v + mhi];
        #pragma unroll
        for (int nt = 0; nt < 4; ++nt) {
            v8f acc = {};
            #pragma unroll
            for (int c = 0; c < 4; ++c)
                acc = __builtin_amdgcn_wmma_f32_16x16x32_f16(
                    false, Af[c], false, Bf[nt * 4 + c],
                    (short)0, acc, false, false);
            float hs = hsrc[nt * 16 + eL][j];
            #pragma unroll
            for (int v = 0; v < 8; ++v)
                macc[nt][v] += (acc[v] + bias[v]) * hs;
        }
    }

    #pragma unroll
    for (int nt = 0; nt < 4; ++nt) {
        int e = e0 + nt * 16 + eL;
        int d = ei[N_EDGES + e];
        #pragma unroll
        for (int v = 0; v < 8; ++v) {
            int o = w * 16 + v + mhi;
            atomicAdd(&agg[d * DIM + o], macc[nt][v]);
        }
    }
}

// ---------------------------------------------------------------------------
// Node update: xc = relu(agg + h@root_w + conv_b); h = GRU(xc, h)
// ---------------------------------------------------------------------------
__global__ __launch_bounds__(64) void k_update(
    const float* __restrict__ agg, const float* __restrict__ h_in,
    const float* __restrict__ root_w, const float* __restrict__ conv_b,
    const float* __restrict__ wih, const float* __restrict__ whh,
    const float* __restrict__ bih, const float* __restrict__ bhh,
    float* __restrict__ h_out) {
    __shared__ float hl[64];
    __shared__ float xc[64];
    const int n = blockIdx.x, t = threadIdx.x;
    hl[t] = h_in[n * DIM + t];
    __syncthreads();
    float a = agg[n * DIM + t] + conv_b[t];
    #pragma unroll 4
    for (int k = 0; k < 64; ++k) a += hl[k] * root_w[k * 64 + t];
    xc[t] = a > 0.f ? a : 0.f;
    __syncthreads();
    float gr = bih[t], gz = bih[64 + t], gn = bih[128 + t];
    float hr = bhh[t], hz = bhh[64 + t], hn = bhh[128 + t];
    #pragma unroll 4
    for (int k = 0; k < 64; ++k) {
        float xk = xc[k], hk = hl[k];
        gr += xk * wih[t * 64 + k];        hr += hk * whh[t * 64 + k];
        gz += xk * wih[(64 + t) * 64 + k]; hz += hk * whh[(64 + t) * 64 + k];
        gn += xk * wih[(128 + t) * 64 + k];hn += hk * whh[(128 + t) * 64 + k];
    }
    float r  = 1.f / (1.f + __expf(-(gr + hr)));
    float z  = 1.f / (1.f + __expf(-(gz + hz)));
    float nn = tanhf(gn + r * hn);
    h_out[n * DIM + t] = (1.f - z) * nn + z * hl[t];
}

// ---------------------------------------------------------------------------
// Global mean pool (atomic accumulation)
// ---------------------------------------------------------------------------
__global__ void k_pool(const float* __restrict__ h, const int* __restrict__ batch,
                       float* __restrict__ pool, float* __restrict__ cnt) {
    int idx = blockIdx.x * 256 + threadIdx.x;
    if (idx >= N_NODES * DIM) return;
    int n = idx >> 6, o = idx & 63;
    int g = batch[n];
    atomicAdd(&pool[g * DIM + o], h[idx]);
    if (o == 0) atomicAdd(&cnt[g], 1.f);
}

// ---------------------------------------------------------------------------
// Output head: relu(out0) -> out1 -> out2
// ---------------------------------------------------------------------------
__global__ __launch_bounds__(64) void k_head(
    const float* __restrict__ pool, const float* __restrict__ cnt,
    const float* __restrict__ w0, const float* __restrict__ b0,
    const float* __restrict__ w1, const float* __restrict__ b1,
    const float* __restrict__ w2, const float* __restrict__ b2,
    float* __restrict__ out) {
    __shared__ float p[64];
    __shared__ float m0[64];
    __shared__ float m1[32];
    const int g = blockIdx.x, t = threadIdx.x;
    float c = cnt[g];
    p[t] = pool[g * DIM + t] / (c > 1.f ? c : 1.f);
    __syncthreads();
    float a = b0[t];
    #pragma unroll 4
    for (int k = 0; k < 64; ++k) a += p[k] * w0[t * 64 + k];
    m0[t] = a > 0.f ? a : 0.f;
    __syncthreads();
    if (t < 32) {
        float s = b1[t];
        #pragma unroll 4
        for (int k = 0; k < 64; ++k) s += m0[k] * w1[t * 64 + k];
        m1[t] = s;
    }
    __syncthreads();
    if (t == 0) {
        float s = b2[0];
        #pragma unroll 4
        for (int k = 0; k < 32; ++k) s += m1[k] * w2[k];
        out[g] = s;
    }
}

// ---------------------------------------------------------------------------
extern "C" void kernel_launch(void* const* d_in, const int* in_sizes, int n_in,
                              void* d_out, int out_size, void* d_ws, size_t ws_size,
                              hipStream_t stream) {
    const float* x         = (const float*)d_in[0];
    const float* edge_attr = (const float*)d_in[1];
    const float* fc0_w     = (const float*)d_in[2];
    const float* fc0_b     = (const float*)d_in[3];
    const float* gru0_wih  = (const float*)d_in[4];
    const float* gru0_bih  = (const float*)d_in[6];
    const float* gru0_bhh  = (const float*)d_in[7];
    const float* mlp1_w    = (const float*)d_in[8];
    const float* mlp1_b    = (const float*)d_in[9];
    const float* mlp2_w    = (const float*)d_in[10];
    const float* mlp2_b    = (const float*)d_in[11];
    const float* root_w    = (const float*)d_in[12];
    const float* conv_b    = (const float*)d_in[13];
    const float* gru_wih   = (const float*)d_in[14];
    const float* gru_whh   = (const float*)d_in[15];
    const float* gru_bih   = (const float*)d_in[16];
    const float* gru_bhh   = (const float*)d_in[17];
    const float* out0_w    = (const float*)d_in[18];
    const float* out0_b    = (const float*)d_in[19];
    const float* out1_w    = (const float*)d_in[20];
    const float* out1_b    = (const float*)d_in[21];
    const float* out2_w    = (const float*)d_in[22];
    const float* out2_b    = (const float*)d_in[23];
    const int*   ei        = (const int*)d_in[24];
    const int*   batch     = (const int*)d_in[25];

    char* ws = (char*)d_ws;
    f16*   packA = (f16*)ws;                                  // 3 MB
    float* hb0   = (float*)(ws + 3 * 1048576);                // 2 MB
    float* hb1   = (float*)(ws + 5 * 1048576);                // 2 MB
    float* agg   = (float*)(ws + 7 * 1048576);                // 2 MB
    float* pool  = (float*)(ws + 9 * 1048576);                // 16 KB
    float* cnt   = (float*)(ws + 9 * 1048576 + 16384);        // 256 B

    // Pack mlp2_w into WMMA A-fragment layout (f16), 1.5M halves.
    {
        int total = DEPTHS * PACKA_HALVES_PER_DEPTH;
        k_pack_mlp2<<<(total + 255) / 256, 256, 0, stream>>>(mlp2_w, packA);
    }

    // Node encoder
    k_node0<<<N_NODES, 64, 0, stream>>>(x, fc0_w, fc0_b, gru0_wih,
                                        gru0_bih, gru0_bhh, hb0);

    float* hbuf[2] = {hb0, hb1};
    for (int d = 0; d < DEPTHS; ++d) {
        float* h_in  = hbuf[d & 1];
        float* h_out = hbuf[(d + 1) & 1];
        hipMemsetAsync(agg, 0, (size_t)N_NODES * DIM * sizeof(float), stream);
        k_edge<<<N_EDGES / 64, 128, 0, stream>>>(
            packA + (size_t)d * PACKA_HALVES_PER_DEPTH,
            mlp1_w + (size_t)d * 128 * 10, mlp1_b + (size_t)d * 128,
            mlp2_b + (size_t)d * 4096, edge_attr, ei, h_in, agg);
        k_update<<<N_NODES, 64, 0, stream>>>(
            agg, h_in, root_w + (size_t)d * DIM * DIM, conv_b + (size_t)d * DIM,
            gru_wih + (size_t)d * 192 * 64, gru_whh + (size_t)d * 192 * 64,
            gru_bih + (size_t)d * 192, gru_bhh + (size_t)d * 192, h_out);
    }
    float* h_final = hbuf[DEPTHS & 1];

    hipMemsetAsync(pool, 0, (size_t)N_GRAPHS * DIM * sizeof(float), stream);
    hipMemsetAsync(cnt, 0, (size_t)N_GRAPHS * sizeof(float), stream);
    k_pool<<<(N_NODES * DIM + 255) / 256, 256, 0, stream>>>(h_final, batch, pool, cnt);
    k_head<<<N_GRAPHS, 64, 0, stream>>>(pool, cnt, out0_w, out0_b,
                                        out1_w, out1_b, out2_w, out2_b,
                                        (float*)d_out);
    (void)in_sizes; (void)n_in; (void)out_size; (void)ws_size;
}
